// USTC_SVS_31095563223404
// MI455X (gfx1250) — matrix operand order repaired and checked by reading, not compiled, not run on hardware
//
#include <hip/hip_runtime.h>
#include <hip/hip_bf16.h>
#include <math.h>

// ---------------------------------------------------------------------------
// Types / WMMA helpers (CDNA5 / gfx1250, wave32)
// ---------------------------------------------------------------------------
typedef __bf16 bf16_t;
typedef __attribute__((ext_vector_type(16))) __bf16 v16bf;
typedef __attribute__((ext_vector_type(8)))  __bf16 v8bf;
typedef __attribute__((ext_vector_type(8)))  float  v8f;

#define DEV static __device__ __forceinline__

DEV v8f wmma_bf16(v16bf a, v16bf b, v8f c) {
  return __builtin_amdgcn_wmma_f32_16x16x32_bf16(false, a, false, b, (short)0, c,
                                                 false, false);
}

union U16 { v16bf v; v8bf h2[2]; };

// A-matrix (16x32 bf16) fragment from row-major [rows][lda] storage.
// ISA layout: lane m=L&15, half=L>>4; element e -> k = 16*(e>>3) + 8*half + (e&7)
// => two contiguous 8-half (16B) chunks per lane.
DEV v16bf load_a(const bf16_t* A, int lda, int row0, int k0, int lane) {
  int m = lane & 15, hh = lane >> 4;
  const bf16_t* p = A + (size_t)(row0 + m) * lda + k0 + 8 * hh;
  U16 u;
  u.h2[0] = *(const v8bf*)p;
  u.h2[1] = *(const v8bf*)(p + 16);
  return u.v;
}

// B-matrix fragment from pre-packed fragment-major buffer: tile*512 + lane*16.
DEV v16bf load_b(const bf16_t* Bp, size_t tile, int lane) {
  return *(const v16bf*)(Bp + tile * 512 + lane * 16);
}

DEV float sigm(float x) { return 1.f / (1.f + __expf(-x)); }

// Cooperative 16-wave GEMM over small LDS-resident A; D via epilogue lambda.
template <typename F>
DEV void wg_gemm(const bf16_t* A, int lda, const bf16_t* Bp, int mtiles,
                 int ktiles, int ntiles, int lane, int wave, F epi) {
  for (int t = wave; t < mtiles * ntiles; t += 16) {
    int mt = t / ntiles, nt = t - mt * ntiles;
    v8f acc = {};
    for (int kt = 0; kt < ktiles; ++kt) {
      v16bf a = load_a(A, lda, mt * 16, kt * 32, lane);
      v16bf b = load_b(Bp, (size_t)kt * ntiles + nt, lane);
      acc = wmma_bf16(a, b, acc);
    }
    int half = lane >> 4, l15 = lane & 15;
#pragma unroll
    for (int r = 0; r < 8; ++r) epi(mt * 16 + r + 8 * half, nt * 16 + l15, acc[r]);
  }
}

// ---------------------------------------------------------------------------
// Model constants
// ---------------------------------------------------------------------------
#define NB_  32     // batch
#define NT_  1000   // frames
#define NF_  1008   // output buffer frames
#define NROWS (NB_ * NT_)

// ---------------------------------------------------------------------------
// Weight packing: f32 [N][ldw] (PyTorch out-major) -> bf16 WMMA B-fragments.
// dst tile order kt-major: tile = kt*ntiles + nt; within tile lane*16 + e;
// element (lane,e) = W[n][koff + k], k = kt*32 + 16*(e>>3) + 8*(lane>>4) + (e&7), n = nt*16+(lane&15)
// ---------------------------------------------------------------------------
__global__ void pack_b_kernel(const float* __restrict__ w, bf16_t* __restrict__ dst,
                              int N, int Ksub, int koff, int ldw, int ktiles,
                              int ntiles) {
  size_t i = (size_t)blockIdx.x * blockDim.x + threadIdx.x;
  size_t total = (size_t)ktiles * ntiles * 512;
  if (i >= total) return;
  int e = (int)(i & 15);
  int lane = (int)((i >> 4) & 31);
  size_t t = i >> 9;
  int nt = (int)(t % ntiles);
  int kt = (int)(t / ntiles);
  int k = kt * 32 + ((e >> 3) << 4) + ((lane >> 4) << 3) + (e & 7);
  int n = nt * 16 + (lane & 15);
  float v = 0.f;
  if (k < Ksub && n < N) v = w[(size_t)n * ldw + koff + k];
  dst[i] = (bf16_t)v;
}

__global__ void zero_f32_kernel(float* __restrict__ p, int n) {
  int i = blockIdx.x * blockDim.x + threadIdx.x;
  if (i < n) p[i] = 0.f;
}

// ---------------------------------------------------------------------------
// Embedding / concat front-end: cat[row][1536] bf16
// ---------------------------------------------------------------------------
__global__ void build_cat_kernel(const int* __restrict__ phone,
                                 const float* __restrict__ pitch,
                                 const int* __restrict__ beats,
                                 const float* __restrict__ embp,
                                 const float* __restrict__ fpw,
                                 const float* __restrict__ fpb,
                                 const float* __restrict__ embb,
                                 bf16_t* __restrict__ cat) {
  size_t idx = (size_t)blockIdx.x * 256 + threadIdx.x;
  if (idx >= (size_t)NROWS * 1536) return;
  size_t row = idx / 1536;
  int col = (int)(idx - row * 1536);
  float v;
  if (col < 512) {
    v = embp[(size_t)phone[row] * 512 + col];
  } else if (col < 1024) {
    int n = col - 512;
    v = fmaxf(pitch[row] * fpw[n] + fpb[n], 0.f);
  } else {
    int n = col - 1024;
    v = embb[(size_t)beats[row] * 512 + n];
  }
  cat[idx] = (bf16_t)v;
}

// ---------------------------------------------------------------------------
// Batched GEMM: D[M=32000][N] = A[M][K](bf16) x Bp(packed) + bias
// mode 0: f32 store; mode 1: tanh -> bf16 store.
// Each wave: one 16-row tile x 4 col-tiles.
// ---------------------------------------------------------------------------
__global__ void __launch_bounds__(256)
gemm_bf16_kernel(const bf16_t* __restrict__ A, const bf16_t* __restrict__ Bp,
                 const float* __restrict__ bias, void* __restrict__ Dv, int K,
                 int N, int mode) {
  const int lane = threadIdx.x & 31;
  const int gw = blockIdx.x * 8 + (threadIdx.x >> 5);
  const int nt4 = N >> 6;
  const int mt = gw / nt4;
  const int ng = gw - mt * nt4;
  const int ntiles = N >> 4;
  const int ktiles = K >> 5;
  v8f z = {};
  v8f acc[4];
  acc[0] = z; acc[1] = z; acc[2] = z; acc[3] = z;
  const bf16_t* Arow = A + (size_t)mt * 16 * K;
  for (int kt = 0; kt < ktiles; ++kt) {
    v16bf a = load_a(Arow, K, 0, kt * 32, lane);
    const bf16_t* bt = Bp + ((size_t)kt * ntiles + ng * 4) * 512 + lane * 16;
#pragma unroll
    for (int jj = 0; jj < 4; ++jj)
      acc[jj] = wmma_bf16(a, *(const v16bf*)(bt + jj * 512), acc[jj]);
  }
  const int half = lane >> 4, l15 = lane & 15;
#pragma unroll
  for (int jj = 0; jj < 4; ++jj) {
    int col = ng * 64 + jj * 16 + l15;
    float bv = bias[col];
#pragma unroll
    for (int r = 0; r < 8; ++r) {
      int row = mt * 16 + r + 8 * half;
      float v = acc[jj][r] + bv;
      if (mode == 1)
        ((bf16_t*)Dv)[(size_t)row * N + col] = (bf16_t)tanhf(v);
      else
        ((float*)Dv)[(size_t)row * N + col] = v;
    }
  }
}

// ---------------------------------------------------------------------------
// Bi-GRU recurrence. block 0 = forward, block 1 = backward. 8 waves.
// Wave w owns hidden columns [w*32, w*32+32) -> its 12 WMMA tiles cover
// exactly the r/z/n pre-activations for those columns (no cross-wave traffic).
// ---------------------------------------------------------------------------
__global__ void __launch_bounds__(256)
bi_scan_kernel(const float* __restrict__ gi_f, const float* __restrict__ gi_b,
               const bf16_t* __restrict__ whhf_p, const bf16_t* __restrict__ whhb_p,
               const float* __restrict__ bhhf, const float* __restrict__ bhhb,
               bf16_t* __restrict__ bi) {
  __shared__ alignas(16) float sh_h[32 * 256];
  __shared__ alignas(16) bf16_t sh_hb[32 * 256];

  const bool bwd = (blockIdx.x == 1);
  const float* gi = bwd ? gi_b : gi_f;
  const bf16_t* wp = bwd ? whhb_p : whhf_p;
  const float* bhh = bwd ? bhhb : bhhf;
  const int cb = bwd ? 256 : 0;

  const int tid = threadIdx.x, lane = tid & 31, w = tid >> 5;
  const int wcol = w * 32;
  for (int k = tid; k < 32 * 256; k += 256) {
    sh_h[k] = 0.f;
    sh_hb[k] = (bf16_t)0.f;
  }
  __syncthreads();

  const v8f z = {};
  for (int step = 0; step < NT_; ++step) {
    const int t = bwd ? (NT_ - 1 - step) : step;
    v8f acc[2][6];
    for (int a1 = 0; a1 < 2; ++a1)
      for (int a2 = 0; a2 < 6; ++a2) acc[a1][a2] = z;

    for (int kt = 0; kt < 8; ++kt) {
      for (int mt = 0; mt < 2; ++mt) {
        v16bf a = load_a(sh_hb, 256, mt * 16, kt * 32, lane);
#pragma unroll
        for (int g = 0; g < 3; ++g)
#pragma unroll
          for (int s = 0; s < 2; ++s) {
            int nt = (g * 256 + wcol + s * 16) >> 4;
            v16bf b = load_b(wp, (size_t)kt * 48 + nt, lane);
            acc[mt][g * 2 + s] = wmma_bf16(a, b, acc[mt][g * 2 + s]);
          }
      }
    }
    __syncthreads();  // all h reads done before h writes

    const int half = lane >> 4, l15 = lane & 15;
    for (int mt = 0; mt < 2; ++mt)
      for (int s = 0; s < 2; ++s) {
#pragma unroll
        for (int r = 0; r < 8; ++r) {
          int brow = mt * 16 + r + 8 * half;
          int j = wcol + s * 16 + l15;
          size_t gir = ((size_t)brow * NT_ + t) * 768;
          float hr = acc[mt][0 + s][r] + bhh[j];
          float hz = acc[mt][2 + s][r] + bhh[256 + j];
          float hn = acc[mt][4 + s][r] + bhh[512 + j];
          float rg = sigm(gi[gir + j] + hr);
          float zg = sigm(gi[gir + 256 + j] + hz);
          float ng = tanhf(gi[gir + 512 + j] + rg * hn);
          float ho = sh_h[brow * 256 + j];
          float hv = (1.f - zg) * ng + zg * ho;
          sh_h[brow * 256 + j] = hv;
          bf16_t hb = (bf16_t)hv;
          sh_hb[brow * 256 + j] = hb;
          bi[((size_t)brow * NT_ + t) * 512 + cb + j] = hb;
        }
      }
    __syncthreads();
  }
}

// ---------------------------------------------------------------------------
// Autoregressive decoder: prenet + uni-GRU cell + fc_lin, persistent 16 waves.
// ---------------------------------------------------------------------------
__global__ void __launch_bounds__(512)
decoder_kernel(const float* __restrict__ giu,       // [32000,384] bi-part + bih
               const float* __restrict__ uni_bhh,   // [384]
               const bf16_t* __restrict__ uwihpn_p, // K=128,N=384
               const bf16_t* __restrict__ uwhh_p,   // K=128,N=384
               const bf16_t* __restrict__ fclin_p,  // K=128,N=80
               const float* __restrict__ fc_lin_b,
               const bf16_t* __restrict__ pfc1_p, const float* __restrict__ p_fc1_b,
               const bf16_t* __restrict__ pfc2_p, const float* __restrict__ p_fc2_b,
               const float* __restrict__ conv_w, const float* __restrict__ conv_b,
               const float* __restrict__ bn_g, const float* __restrict__ bn_b,
               const float* __restrict__ bn_m, const float* __restrict__ bn_v,
               const bf16_t* __restrict__ fpos_p, const float* __restrict__ fc_pos_b,
               const bf16_t* __restrict__ attq_p, const float* __restrict__ att_q_b,
               const bf16_t* __restrict__ attk_p, const float* __restrict__ att_k_b,
               const bf16_t* __restrict__ attv_p, const float* __restrict__ att_v_b,
               const bf16_t* __restrict__ atto_p, const float* __restrict__ att_o_b,
               const bf16_t* __restrict__ pfc3_p, const float* __restrict__ p_fc3_b,
               float* __restrict__ out) {
  __shared__ alignas(16) float sh_h[32 * 128];
  __shared__ alignas(16) bf16_t sh_hb[32 * 128];
  __shared__ alignas(16) unsigned char sb[32768];  // phased scratch

  bf16_t* s_x   = (bf16_t*)sb;             // [96][96]  prenet input (padded K)
  bf16_t* s_h1  = (bf16_t*)(sb + 18432);   // [96][64]
  bf16_t* s_h2  = (bf16_t*)sb;             // [96][64]  (aliases s_x)
  float*  s_cf  = (float*)(sb + 12288);    // [64][64]  f32 c
  bf16_t* s_cpe = (bf16_t*)sb;             // [64][64]
  bf16_t* s_cb  = (bf16_t*)sb;             // [64][64]
  bf16_t* s_q   = (bf16_t*)(sb + 8192);    // [64][64]
  bf16_t* s_k   = (bf16_t*)(sb + 16384);   // [64][64]
  bf16_t* s_v   = (bf16_t*)(sb + 24576);   // [64][64]
  bf16_t* s_pn  = (bf16_t*)(sb + 16384);   // [32][128] (aliases s_k, later)

  const int tid = threadIdx.x;
  const int lane = tid & 31, wave = tid >> 5;

  for (int k = tid; k < 32 * 128; k += 512) {
    sh_h[k] = 0.f;
    sh_hb[k] = (bf16_t)0.f;
  }
  __syncthreads();

  for (int i = 0; i < NT_; ++i) {
    // ---- A: prenet input x = pad(hist,(0,1)) as bf16 [96][96] (K padded) ----
    {
      int start = (i < 2) ? 0 : i;  // faithful to the torch loop
      for (int idx = tid; idx < 96 * 96; idx += 512) {
        int row = idx / 96, col = idx - row * 96;
        int b = row / 3, j = row - b * 3;
        float v = 0.f;
        if (j < 2 && col < 80) v = out[((size_t)b * NF_ + start + j) * 80 + col];
        s_x[idx] = (bf16_t)v;
      }
    }
    __syncthreads();
    // ---- B: p_fc1 + relu ----
    wg_gemm(s_x, 96, pfc1_p, 6, 3, 4, lane, wave, [&](int r, int c, float v) {
      s_h1[r * 64 + c] = (bf16_t)fmaxf(v + p_fc1_b[c], 0.f);
    });
    __syncthreads();
    // ---- C: p_fc2 + relu ----
    wg_gemm(s_h1, 64, pfc2_p, 6, 2, 4, lane, wave, [&](int r, int c, float v) {
      s_h2[r * 64 + c] = (bf16_t)fmaxf(v + p_fc2_b[c], 0.f);
    });
    __syncthreads();
    // ---- D: conv1d k=2 + batchnorm + relu -> s_cf ----
    for (int idx = tid; idx < 64 * 64; idx += 512) {
      int row = idx >> 6, n = idx & 63;
      int b = row >> 1, j = row & 1;
      const bf16_t* h0 = s_h2 + (b * 3 + j) * 64;
      const bf16_t* h1r = h0 + 64;
      float s = conv_b[n];
      const float* wr = conv_w + n * 128;
      for (int k = 0; k < 64; ++k)
        s += (float)h0[k] * wr[k * 2] + (float)h1r[k] * wr[k * 2 + 1];
      s = (s - bn_m[n]) * rsqrtf(bn_v[n] + 1e-5f) * bn_g[n] + bn_b[n];
      s_cf[idx] = fmaxf(s, 0.f);
    }
    __syncthreads();
    // ---- D2: c + sinusoidal PE -> bf16 ----
    for (int idx = tid; idx < 64 * 64; idx += 512) {
      int row = idx >> 6, n = idx & 63;
      int j = row & 1;
      float freq = __expf((float)(n & ~1) * (-9.210340371976184f / 64.f));
      float pe = (n & 1) ? __cosf((float)j * freq) : __sinf((float)j * freq);
      s_cpe[idx] = (bf16_t)(s_cf[idx] + pe);
    }
    __syncthreads();
    // ---- E: c = c + fc_pos(c + pe) ----
    wg_gemm(s_cpe, 64, fpos_p, 4, 2, 4, lane, wave,
            [&](int r, int c, float v) { s_cf[r * 64 + c] += v + fc_pos_b[c]; });
    __syncthreads();
    // ---- F0: master c -> bf16 ----
    for (int idx = tid; idx < 64 * 64; idx += 512) s_cb[idx] = (bf16_t)s_cf[idx];
    __syncthreads();
    // ---- F: 3 attention blocks ----
    for (int blk = 0; blk < 3; ++blk) {
      const bf16_t* qp = attq_p + blk * 4096; const float* qb = att_q_b + blk * 64;
      const bf16_t* kp = attk_p + blk * 4096; const float* kb = att_k_b + blk * 64;
      const bf16_t* vp = attv_p + blk * 4096; const float* vb = att_v_b + blk * 64;
      const bf16_t* op = atto_p + blk * 4096; const float* ob = att_o_b + blk * 64;
      wg_gemm(s_cb, 64, qp, 4, 2, 4, lane, wave, [&](int r, int c, float v) {
        s_q[r * 64 + c] = (bf16_t)(v + qb[c]);
      });
      wg_gemm(s_cb, 64, kp, 4, 2, 4, lane, wave, [&](int r, int c, float v) {
        s_k[r * 64 + c] = (bf16_t)(v + kb[c]);
      });
      wg_gemm(s_cb, 64, vp, 4, 2, 4, lane, wave, [&](int r, int c, float v) {
        s_v[r * 64 + c] = (bf16_t)(v + vb[c]);
      });
      __syncthreads();
      if (tid < 128) {  // (b,h,qj): scores, softmax(2), a@v -> own q slice
        int b = tid >> 2, hh = (tid >> 1) & 1, qj = tid & 1;
        const bf16_t* qr = s_q + (b * 2 + qj) * 64 + hh * 32;
        float s0 = 0.f, s1 = 0.f;
        for (int d = 0; d < 32; ++d) {
          float qv = (float)qr[d];
          s0 += qv * (float)s_k[(b * 2 + 0) * 64 + hh * 32 + d];
          s1 += qv * (float)s_k[(b * 2 + 1) * 64 + hh * 32 + d];
        }
        const float scl = 0.1767766952966369f;  // 1/sqrt(32)
        s0 *= scl; s1 *= scl;
        float mx = fmaxf(s0, s1);
        float a0 = __expf(s0 - mx), a1 = __expf(s1 - mx);
        float inv = 1.f / (a0 + a1);
        a0 *= inv; a1 *= inv;
        for (int d = 0; d < 32; ++d) {
          float ov = a0 * (float)s_v[(b * 2 + 0) * 64 + hh * 32 + d] +
                     a1 * (float)s_v[(b * 2 + 1) * 64 + hh * 32 + d];
          s_q[(b * 2 + qj) * 64 + hh * 32 + d] = (bf16_t)ov;
        }
      }
      __syncthreads();
      wg_gemm(s_q, 64, op, 4, 2, 4, lane, wave, [&](int r, int c, float v) {
        s_cb[r * 64 + c] = (bf16_t)(v + ob[c]);
      });
      __syncthreads();
    }
    // ---- G: pn = c + p_fc3(c), reshape [64][64] -> [32][128] ----
    wg_gemm(s_cb, 64, pfc3_p, 4, 2, 4, lane, wave, [&](int r, int c, float v) {
      float pv = (float)s_cb[r * 64 + c] + v + p_fc3_b[c];
      s_pn[(r >> 1) * 128 + (r & 1) * 64 + c] = (bf16_t)pv;
    });
    __syncthreads();
    // ---- H: gi_pn + gh GEMMs + GRU cell. Wave owns (mt, jt) column tile. ----
    {
      int jt = wave & 7, mt = wave >> 3;
      v8f aR = {}, aZ = {}, aNgi = {}, aNgh = {};
      for (int kt = 0; kt < 4; ++kt) {
        v16bf apn = load_a(s_pn, 128, mt * 16, kt * 32, lane);
        v16bf ah = load_a(sh_hb, 128, mt * 16, kt * 32, lane);
        v16bf b;
        b = load_b(uwihpn_p, (size_t)kt * 24 + jt, lane);      aR = wmma_bf16(apn, b, aR);
        b = load_b(uwhh_p,   (size_t)kt * 24 + jt, lane);      aR = wmma_bf16(ah, b, aR);
        b = load_b(uwihpn_p, (size_t)kt * 24 + 8 + jt, lane);  aZ = wmma_bf16(apn, b, aZ);
        b = load_b(uwhh_p,   (size_t)kt * 24 + 8 + jt, lane);  aZ = wmma_bf16(ah, b, aZ);
        b = load_b(uwihpn_p, (size_t)kt * 24 + 16 + jt, lane); aNgi = wmma_bf16(apn, b, aNgi);
        b = load_b(uwhh_p,   (size_t)kt * 24 + 16 + jt, lane); aNgh = wmma_bf16(ah, b, aNgh);
      }
      __syncthreads();  // all sh_hb reads done before h writes
      int half = lane >> 4, l15 = lane & 15;
#pragma unroll
      for (int r = 0; r < 8; ++r) {
        int brow = mt * 16 + r + 8 * half;
        int j = jt * 16 + l15;
        const float* g = giu + ((size_t)brow * NT_ + i) * 384;
        float rg = sigm(aR[r] + g[j] + uni_bhh[j]);
        float zg = sigm(aZ[r] + g[128 + j] + uni_bhh[128 + j]);
        float hn = aNgh[r] + uni_bhh[256 + j];
        float ng = tanhf(aNgi[r] + g[256 + j] + rg * hn);
        float ho = sh_h[brow * 128 + j];
        float hv = (1.f - zg) * ng + zg * ho;
        sh_h[brow * 128 + j] = hv;
        sh_hb[brow * 128 + j] = (bf16_t)hv;
      }
    }
    __syncthreads();
    // ---- Y: y = h @ fc_lin^T + b -> out frame i ----
    if (wave < 10) {
      int mt = wave / 5, nt = wave - mt * 5;
      v8f acc = {};
      for (int kt = 0; kt < 4; ++kt) {
        v16bf a = load_a(sh_hb, 128, mt * 16, kt * 32, lane);
        v16bf b = load_b(fclin_p, (size_t)kt * 5 + nt, lane);
        acc = wmma_bf16(a, b, acc);
      }
      int half = lane >> 4, l15 = lane & 15;
#pragma unroll
      for (int r = 0; r < 8; ++r) {
        int brow = mt * 16 + r + 8 * half;
        int c = nt * 16 + l15;
        out[((size_t)brow * NF_ + i) * 80 + c] = acc[r] + fc_lin_b[c];
      }
    }
    __threadfence_block();
    __syncthreads();
  }
}

// ---------------------------------------------------------------------------
// Host side
// ---------------------------------------------------------------------------
extern "C" void kernel_launch(void* const* d_in, const int* in_sizes, int n_in,
                              void* d_out, int out_size, void* d_ws,
                              size_t ws_size, hipStream_t stream) {
  (void)in_sizes; (void)n_in; (void)out_size; (void)ws_size;

  const int*   phone     = (const int*)d_in[0];
  const float* pitch     = (const float*)d_in[1];
  const int*   beats     = (const int*)d_in[2];
  // d_in[3] length, d_in[4] num_frames: unused (all lengths == T, NF fixed)
  const float* emb_phone = (const float*)d_in[5];
  const float* fc_pitch_w= (const float*)d_in[6];
  const float* fc_pitch_b= (const float*)d_in[7];
  const float* emb_beats = (const float*)d_in[8];
  const float* fc1_w = (const float*)d_in[9];
  const float* fc1_b = (const float*)d_in[10];
  const float* fc2_w = (const float*)d_in[11];
  const float* fc2_b = (const float*)d_in[12];
  const float* bi_wih_f = (const float*)d_in[13];
  const float* bi_whh_f = (const float*)d_in[14];
  const float* bi_bih_f = (const float*)d_in[15];
  const float* bi_bhh_f = (const float*)d_in[16];
  const float* bi_wih_b = (const float*)d_in[17];
  const float* bi_whh_b = (const float*)d_in[18];
  const float* bi_bih_b = (const float*)d_in[19];
  const float* bi_bhh_b = (const float*)d_in[20];
  const float* uni_wih = (const float*)d_in[21];
  const float* uni_whh = (const float*)d_in[22];
  const float* uni_bih = (const float*)d_in[23];
  const float* uni_bhh = (const float*)d_in[24];
  const float* fc_lin_w = (const float*)d_in[25];
  const float* fc_lin_b = (const float*)d_in[26];
  const float* p_fc1_w = (const float*)d_in[27];
  const float* p_fc1_b = (const float*)d_in[28];
  const float* p_fc2_w = (const float*)d_in[29];
  const float* p_fc2_b = (const float*)d_in[30];
  const float* conv_w = (const float*)d_in[31];
  const float* conv_b = (const float*)d_in[32];
  const float* bn_g = (const float*)d_in[33];
  const float* bn_b = (const float*)d_in[34];
  const float* bn_m = (const float*)d_in[35];
  const float* bn_v = (const float*)d_in[36];
  const float* fc_pos_w = (const float*)d_in[37];
  const float* fc_pos_b = (const float*)d_in[38];
  const float* att_q_w = (const float*)d_in[39];
  const float* att_q_b = (const float*)d_in[40];
  const float* att_k_w = (const float*)d_in[41];
  const float* att_k_b = (const float*)d_in[42];
  const float* att_v_w = (const float*)d_in[43];
  const float* att_v_b = (const float*)d_in[44];
  const float* att_o_w = (const float*)d_in[45];
  const float* att_o_b = (const float*)d_in[46];
  const float* p_fc3_w = (const float*)d_in[47];
  const float* p_fc3_b = (const float*)d_in[48];

  unsigned char* ws = (unsigned char*)d_ws;
  size_t off = 0;
  auto take = [&](size_t bytes) -> unsigned char* {
    unsigned char* p = ws + off;
    off = (off + bytes + 255) & ~(size_t)255;
    return p;
  };

  // Activations (cat reused as gi_f after fc1 has consumed it: same 98.3 MB)
  bf16_t* cat  = (bf16_t*)take((size_t)NROWS * 1536 * 2);
  float*  gi_f = (float*)cat;
  bf16_t* h1   = (bf16_t*)take((size_t)NROWS * 512 * 2);
  bf16_t* xb   = (bf16_t*)take((size_t)NROWS * 512 * 2);
  float*  gi_b = (float*)take((size_t)NROWS * 768 * 4);
  bf16_t* bi   = (bf16_t*)take((size_t)NROWS * 512 * 2);
  float*  giu  = (float*)take((size_t)NROWS * 384 * 4);

  // Packed bf16 weight fragments
  bf16_t* fc1p    = (bf16_t*)take((size_t)48 * 32 * 512 * 2);
  bf16_t* fc2p    = (bf16_t*)take((size_t)16 * 32 * 512 * 2);
  bf16_t* wihf_p  = (bf16_t*)take((size_t)16 * 48 * 512 * 2);
  bf16_t* wihb_p  = (bf16_t*)take((size_t)16 * 48 * 512 * 2);
  bf16_t* whhf_p  = (bf16_t*)take((size_t)8 * 48 * 512 * 2);
  bf16_t* whhb_p  = (bf16_t*)take((size_t)8 * 48 * 512 * 2);
  bf16_t* uwihbi_p= (bf16_t*)take((size_t)16 * 24 * 512 * 2);
  bf16_t* uwihpn_p= (bf16_t*)take((size_t)4 * 24 * 512 * 2);
  bf16_t* uwhh_p  = (bf16_t*)take((size_t)4 * 24 * 512 * 2);
  bf16_t* fclin_p = (bf16_t*)take((size_t)4 * 5 * 512 * 2);
  bf16_t* pfc1_p  = (bf16_t*)take((size_t)3 * 4 * 512 * 2);
  bf16_t* pfc2_p  = (bf16_t*)take((size_t)2 * 4 * 512 * 2);
  bf16_t* fpos_p  = (bf16_t*)take((size_t)2 * 4 * 512 * 2);
  bf16_t* pfc3_p  = (bf16_t*)take((size_t)2 * 4 * 512 * 2);
  bf16_t* attq_p  = (bf16_t*)take((size_t)3 * 8 * 512 * 2);
  bf16_t* attk_p  = (bf16_t*)take((size_t)3 * 8 * 512 * 2);
  bf16_t* attv_p  = (bf16_t*)take((size_t)3 * 8 * 512 * 2);
  bf16_t* atto_p  = (bf16_t*)take((size_t)3 * 8 * 512 * 2);

  auto pack = [&](const float* w, bf16_t* dst, int N, int Ksub, int koff, int ldw,
                  int kt, int nt) {
    size_t total = (size_t)kt * nt * 512;
    int blocks = (int)((total + 255) / 256);
    pack_b_kernel<<<blocks, 256, 0, stream>>>(w, dst, N, Ksub, koff, ldw, kt, nt);
  };

  // Zero the output buffer (decoder reads "future" frames per the torch loop).
  zero_f32_kernel<<<(NB_ * NF_ * 80 + 255) / 256, 256, 0, stream>>>((float*)d_out,
                                                                    NB_ * NF_ * 80);

  // Weight packing
  pack(fc1_w, fc1p, 512, 1536, 0, 1536, 48, 32);
  pack(fc2_w, fc2p, 512, 512, 0, 512, 16, 32);
  pack(bi_wih_f, wihf_p, 768, 512, 0, 512, 16, 48);
  pack(bi_wih_b, wihb_p, 768, 512, 0, 512, 16, 48);
  pack(bi_whh_f, whhf_p, 768, 256, 0, 256, 8, 48);
  pack(bi_whh_b, whhb_p, 768, 256, 0, 256, 8, 48);
  pack(uni_wih, uwihbi_p, 384, 512, 0, 640, 16, 24);
  pack(uni_wih, uwihpn_p, 384, 128, 512, 640, 4, 24);
  pack(uni_whh, uwhh_p, 384, 128, 0, 128, 4, 24);
  pack(fc_lin_w, fclin_p, 80, 128, 0, 128, 4, 5);
  pack(p_fc1_w, pfc1_p, 64, 80, 0, 80, 3, 4);
  pack(p_fc2_w, pfc2_p, 64, 64, 0, 64, 2, 4);
  pack(fc_pos_w, fpos_p, 64, 64, 0, 64, 2, 4);
  pack(p_fc3_w, pfc3_p, 64, 64, 0, 64, 2, 4);
  for (int blk = 0; blk < 3; ++blk) {
    pack(att_q_w + blk * 4096, attq_p + blk * 4096, 64, 64, 0, 64, 2, 4);
    pack(att_k_w + blk * 4096, attk_p + blk * 4096, 64, 64, 0, 64, 2, 4);
    pack(att_v_w + blk * 4096, attv_p + blk * 4096, 64, 64, 0, 64, 2, 4);
    pack(att_o_w + blk * 4096, atto_p + blk * 4096, 64, 64, 0, 64, 2, 4);
  }

  // Front-end
  build_cat_kernel<<<(int)(((size_t)NROWS * 1536 + 255) / 256), 256, 0, stream>>>(
      phone, pitch, beats, emb_phone, fc_pitch_w, fc_pitch_b, emb_beats, cat);
  // fc1 (tanh, bf16), fc2 (tanh, bf16)
  gemm_bf16_kernel<<<2000, 256, 0, stream>>>(cat, fc1p, fc1_b, h1, 1536, 512, 1);
  gemm_bf16_kernel<<<2000, 256, 0, stream>>>(h1, fc2p, fc2_b, xb, 512, 512, 1);
  // Bi-GRU input GEMMs (f32 gi, gi_f aliases cat — cat is dead after fc1)
  gemm_bf16_kernel<<<3000, 256, 0, stream>>>(xb, wihf_p, bi_bih_f, gi_f, 512, 768, 0);
  gemm_bf16_kernel<<<3000, 256, 0, stream>>>(xb, wihb_p, bi_bih_b, gi_b, 512, 768, 0);
  // Persistent bi-GRU recurrences (block 0 fwd, block 1 bwd)
  bi_scan_kernel<<<2, 256, 0, stream>>>(gi_f, gi_b, whhf_p, whhb_p, bi_bhh_f,
                                        bi_bhh_b, bi);
  // Precompute uni-GRU input GEMM for the bi-output part (incl. uni_bih)
  gemm_bf16_kernel<<<1500, 256, 0, stream>>>(bi, uwihbi_p, uni_bih, giu, 512, 384, 0);
  // Autoregressive decoder (persistent, 16 waves)
  decoder_kernel<<<1, 512, 0, stream>>>(
      giu, uni_bhh, uwihpn_p, uwhh_p, fclin_p, fc_lin_b, pfc1_p, p_fc1_b, pfc2_p,
      p_fc2_b, conv_w, conv_b, bn_g, bn_b, bn_m, bn_v, fpos_p, fc_pos_b, attq_p,
      att_q_b, attk_p, att_k_b, attv_p, att_v_b, atto_p, att_o_b, pfc3_p, p_fc3_b,
      (float*)d_out);
}
